// CausalSelfAttention_12618613916273
// MI455X (gfx1250) — compile-verified, matrix operand order checked by reading, not run on hardware
//
#include <hip/hip_runtime.h>

#define D_EMB 1024
#define NHEAD 16
#define HDIM  64
#define TSEQ  2048
#define BATCH 4
#define BT    (BATCH * TSEQ)

typedef __attribute__((ext_vector_type(16))) _Float16 v16h;
typedef __attribute__((ext_vector_type(8)))  _Float16 v8h;
typedef __attribute__((ext_vector_type(8)))  float    v8f;
typedef __attribute__((ext_vector_type(4)))  float    v4f;

#define WMMA_F16(A, B, C) \
  __builtin_amdgcn_wmma_f32_16x16x32_f16(false, (A), false, (B), (short)0, (C), false, false)

// Load a 16x32 f16 WMMA A/B fragment from a row-major [rows x K] matrix.
// Per CDNA5 ISA 16-bit A layout: lane holds row (lane&15); halves 0..7 are
// K = c..c+7 and halves 8..15 are K = 16+c..16+c+7 with c = (lane>=16)*8.
// Both halves are 16-byte contiguous loads.
__device__ __forceinline__ v16h load_frag16(const _Float16* __restrict__ base, int ld) {
  const int lane = threadIdx.x & 31;
  const int row  = lane & 15;
  const int c    = (lane >> 4) << 3;       // 0 or 8
  const _Float16* p = base + (size_t)row * ld + c;
  v8h lo = *reinterpret_cast<const v8h*>(p);
  v8h hi = *reinterpret_cast<const v8h*>(p + 16);
  v16h r;
#pragma unroll
  for (int i = 0; i < 8; ++i) { r[i] = lo[i]; r[8 + i] = hi[i]; }
  return r;
}

__device__ __forceinline__ float rowmax16(float v) {
#pragma unroll
  for (int m = 1; m < 16; m <<= 1) v = fmaxf(v, __shfl_xor(v, m, 32));
  return v;
}
__device__ __forceinline__ float rowsum16(float v) {
#pragma unroll
  for (int m = 1; m < 16; m <<= 1) v += __shfl_xor(v, m, 32);
  return v;
}

// ---------------------------------------------------------------- convert
// 8 elements per thread: two float4 loads -> one 16B f16 store.
__global__ void cvt_f32_to_f16(const float* __restrict__ src,
                               _Float16* __restrict__ dst, int n8) {
  int i = blockIdx.x * blockDim.x + threadIdx.x;
  if (i >= n8) return;
  v4f a = *reinterpret_cast<const v4f*>(src + (size_t)i * 8);
  v4f b = *reinterpret_cast<const v4f*>(src + (size_t)i * 8 + 4);
  v8h o;
#pragma unroll
  for (int j = 0; j < 4; ++j) { o[j] = (_Float16)a[j]; o[4 + j] = (_Float16)b[j]; }
  *reinterpret_cast<v8h*>(dst + (size_t)i * 8) = o;
}

// ---------------------------------------------------------------- QKV GEMM
// q = x @ Wq^T etc.  One wave -> one 16x16 tile, A fragment shared by 3 WMMAs.
// Software-pipelined: fragments for step k+32 are loaded before the WMMAs of
// step k, so the compiler can issue partial loadcnt waits and overlap.
__global__ __launch_bounds__(256) void qkv_kernel(
    const _Float16* __restrict__ xh,
    const _Float16* __restrict__ wq, const _Float16* __restrict__ wk,
    const _Float16* __restrict__ wv,
    _Float16* __restrict__ Q, _Float16* __restrict__ K,
    _Float16* __restrict__ Vt) {
  const int wave = blockIdx.x * (blockDim.x >> 5) + (threadIdx.x >> 5);
  const int mi = wave >> 6;              // 512 M-tiles
  const int ni = wave & 63;              // 64  N-tiles
  const int i0 = mi << 4, j0 = ni << 4;
  const int lane = threadIdx.x & 31;

  const _Float16* pa = xh + (size_t)i0 * D_EMB;
  const _Float16* pq = wq + (size_t)j0 * D_EMB;
  const _Float16* pk = wk + (size_t)j0 * D_EMB;
  const _Float16* pv = wv + (size_t)j0 * D_EMB;

  v8f accq = {}, acck = {}, accv = {};
  v16h a  = load_frag16(pa, D_EMB);
  v16h bq = load_frag16(pq, D_EMB);
  v16h bk = load_frag16(pk, D_EMB);
  v16h bv = load_frag16(pv, D_EMB);
#pragma unroll 4
  for (int k0 = 0; k0 < D_EMB; k0 += 32) {
    v16h an, bqn, bkn, bvn;
    if (k0 + 32 < D_EMB) {
      an  = load_frag16(pa + k0 + 32, D_EMB);
      bqn = load_frag16(pq + k0 + 32, D_EMB);
      bkn = load_frag16(pk + k0 + 32, D_EMB);
      bvn = load_frag16(pv + k0 + 32, D_EMB);
    }
    accq = WMMA_F16(a, bq, accq);
    acck = WMMA_F16(a, bk, acck);
    accv = WMMA_F16(a, bv, accv);
    a = an; bq = bqn; bk = bkn; bv = bvn;
  }

  const int n = lane & 15;
  const int rbase = (lane >> 4) << 3;
  const int col = j0 + n;
  const int h = col >> 6, hd = col & 63;
#pragma unroll
  for (int r = 0; r < 8; ++r) {
    const int row = i0 + rbase + r;               // b*T + t
    const int b = row >> 11, t = row & (TSEQ - 1);
    const size_t qidx = (((size_t)b * NHEAD + h) * TSEQ + t) * HDIM + hd;
    Q[qidx] = (_Float16)accq[r];
    K[qidx] = (_Float16)acck[r];
    const size_t vidx = (((size_t)b * NHEAD + h) * HDIM + hd) * TSEQ + t;
    Vt[vidx] = (_Float16)accv[r];
  }
}

// ---------------------------------------------------------------- attention
// One wave per (b, h, 16-query tile); 32 keys per iteration (online softmax).
// K-fragments for the next key block are prefetched before the softmax, so
// exp/shfl VALU work hides their latency.
__global__ __launch_bounds__(256) void attn_kernel(
    const _Float16* __restrict__ Q, const _Float16* __restrict__ K,
    const _Float16* __restrict__ Vt, _Float16* __restrict__ Y) {
  __shared__ __align__(16) _Float16 pbuf[8][16 * 32];
  const int wslot = threadIdx.x >> 5;
  const int wave = blockIdx.x * 8 + wslot;
  const int qt = wave & 127;                 // 128 query tiles
  const int bh = wave >> 7;
  const int h = bh & 15, b = bh >> 4;
  const int lane = threadIdx.x & 31;
  const int n = lane & 15;
  const int rbase = (lane >> 4) << 3;

  const _Float16* Qb = Q  + ((size_t)b * NHEAD + h) * TSEQ * HDIM;
  const _Float16* Kb = K  + ((size_t)b * NHEAD + h) * TSEQ * HDIM;
  const _Float16* Vb = Vt + ((size_t)b * NHEAD + h) * HDIM * TSEQ;

  const int q0 = qt << 4;
  const v16h qa0 = load_frag16(Qb + (size_t)q0 * HDIM + 0,  HDIM);
  const v16h qa1 = load_frag16(Qb + (size_t)q0 * HDIM + 32, HDIM);

  v8f o0 = {}, o1 = {}, o2 = {}, o3 = {};
  float mrow[8], lrow[8];
#pragma unroll
  for (int r = 0; r < 8; ++r) { mrow[r] = -1e30f; lrow[r] = 0.0f; }

  const float scale = 0.125f;                // 1/sqrt(64)
  const int nkeys = q0 + 16;                 // causal: keys 0..q0+15

  v16h kf0a = load_frag16(Kb + 0,                     HDIM);
  v16h kf0b = load_frag16(Kb + 32,                    HDIM);
  v16h kf1a = load_frag16(Kb + (size_t)16 * HDIM + 0,  HDIM);
  v16h kf1b = load_frag16(Kb + (size_t)16 * HDIM + 32, HDIM);

  for (int kb = 0; kb < nkeys; kb += 32) {
    // S = Q @ K^T for 32 keys (two 16-key column tiles)
    v8f s0 = {}, s1 = {};
    s0 = WMMA_F16(qa0, kf0a, s0);
    s0 = WMMA_F16(qa1, kf0b, s0);
    s1 = WMMA_F16(qa0, kf1a, s1);
    s1 = WMMA_F16(qa1, kf1b, s1);

    // prefetch next key block's K fragments (covered by softmax VALU below)
    if (kb + 32 < nkeys) {
      const _Float16* kn = Kb + (size_t)(kb + 32) * HDIM;
      kf0a = load_frag16(kn + 0,                     HDIM);
      kf0b = load_frag16(kn + 32,                    HDIM);
      kf1a = load_frag16(kn + (size_t)16 * HDIM + 0,  HDIM);
      kf1b = load_frag16(kn + (size_t)16 * HDIM + 32, HDIM);
    }
    // V fragments for this block are independent of the softmax
    v16h vf0 = load_frag16(Vb + (size_t)0  * TSEQ + kb, TSEQ);
    v16h vf1 = load_frag16(Vb + (size_t)16 * TSEQ + kb, TSEQ);
    v16h vf2 = load_frag16(Vb + (size_t)32 * TSEQ + kb, TSEQ);
    v16h vf3 = load_frag16(Vb + (size_t)48 * TSEQ + kb, TSEQ);

    float p0[8], p1[8], alpha[8];
#pragma unroll
    for (int r = 0; r < 8; ++r) {
      const int qg = q0 + rbase + r;
      const float v0 = (kb + n      <= qg) ? s0[r] * scale : -1e30f;
      const float v1 = (kb + 16 + n <= qg) ? s1[r] * scale : -1e30f;
      const float mx = rowmax16(fmaxf(v0, v1));
      const float mnew = fmaxf(mrow[r], mx);
      alpha[r] = __expf(mrow[r] - mnew);
      p0[r] = __expf(v0 - mnew);
      p1[r] = __expf(v1 - mnew);
      const float rs = rowsum16(p0[r] + p1[r]);
      lrow[r] = lrow[r] * alpha[r] + rs;
      mrow[r] = mnew;
    }
    // rescale running output
#pragma unroll
    for (int r = 0; r < 8; ++r) {
      o0[r] *= alpha[r]; o1[r] *= alpha[r]; o2[r] *= alpha[r]; o3[r] *= alpha[r];
    }
    // stage P (16x32, f16) in LDS, re-read as A fragment
    _Float16* pb = &pbuf[wslot][0];
#pragma unroll
    for (int r = 0; r < 8; ++r) {
      pb[(rbase + r) * 32 + n]      = (_Float16)p0[r];
      pb[(rbase + r) * 32 + 16 + n] = (_Float16)p1[r];
    }
    asm volatile("s_wait_dscnt 0" ::: "memory");
    const v16h pf = load_frag16(pb, 32);

    // O += P @ V  (V transposed: row d contiguous along keys)
    o0 = WMMA_F16(pf, vf0, o0);
    o1 = WMMA_F16(pf, vf1, o1);
    o2 = WMMA_F16(pf, vf2, o2);
    o3 = WMMA_F16(pf, vf3, o3);
  }

  // normalize and write y[b, t, h*64 + d] as f16
#pragma unroll
  for (int r = 0; r < 8; ++r) {
    const float inv = 1.0f / lrow[r];
    const int t = q0 + rbase + r;
    _Float16* yrow = Y + ((size_t)b * TSEQ + t) * D_EMB + h * HDIM;
    yrow[0  + n] = (_Float16)(o0[r] * inv);
    yrow[16 + n] = (_Float16)(o1[r] * inv);
    yrow[32 + n] = (_Float16)(o2[r] * inv);
    yrow[48 + n] = (_Float16)(o3[r] * inv);
  }
}

// ---------------------------------------------------------------- out proj
__global__ __launch_bounds__(256) void proj_kernel(
    const _Float16* __restrict__ yh, const _Float16* __restrict__ wp,
    const float* __restrict__ bp, float* __restrict__ out) {
  const int wave = blockIdx.x * (blockDim.x >> 5) + (threadIdx.x >> 5);
  const int mi = wave >> 6;
  const int ni = wave & 63;
  const int i0 = mi << 4, j0 = ni << 4;
  const int lane = threadIdx.x & 31;
  const int n = lane & 15;
  const int rbase = (lane >> 4) << 3;

  const _Float16* pa = yh + (size_t)i0 * D_EMB;
  const _Float16* pw = wp + (size_t)j0 * D_EMB;

  const float bias = bp[j0 + n];
  v8f acc = {};
#pragma unroll
  for (int r = 0; r < 8; ++r) acc[r] = bias;

  v16h a = load_frag16(pa, D_EMB);
  v16h b = load_frag16(pw, D_EMB);
#pragma unroll 4
  for (int k0 = 0; k0 < D_EMB; k0 += 32) {
    v16h an, bn;
    if (k0 + 32 < D_EMB) {
      an = load_frag16(pa + k0 + 32, D_EMB);
      bn = load_frag16(pw + k0 + 32, D_EMB);
    }
    acc = WMMA_F16(a, b, acc);
    a = an; b = bn;
  }
#pragma unroll
  for (int r = 0; r < 8; ++r)
    out[(size_t)(i0 + rbase + r) * D_EMB + j0 + n] = acc[r];
}

// ---------------------------------------------------------------- launch
extern "C" void kernel_launch(void* const* d_in, const int* in_sizes, int n_in,
                              void* d_out, int out_size, void* d_ws, size_t ws_size,
                              hipStream_t stream) {
  const float* x  = (const float*)d_in[0];
  const float* Wq = (const float*)d_in[1];
  const float* Wk = (const float*)d_in[2];
  const float* Wv = (const float*)d_in[3];
  const float* Wp = (const float*)d_in[4];
  const float* bp = (const float*)d_in[5];
  float* out = (float*)d_out;

  char* ws = (char*)d_ws;
  const size_t szX = (size_t)BT * D_EMB * sizeof(_Float16);       // 16 MB
  const size_t szW = (size_t)D_EMB * D_EMB * sizeof(_Float16);    // 2 MB
  size_t off = 0;
  _Float16* xh  = (_Float16*)(ws + off); off += szX;
  _Float16* wqh = (_Float16*)(ws + off); off += szW;
  _Float16* wkh = (_Float16*)(ws + off); off += szW;
  _Float16* wvh = (_Float16*)(ws + off); off += szW;
  _Float16* wph = (_Float16*)(ws + off); off += szW;
  _Float16* Qh  = (_Float16*)(ws + off); off += szX;              // [B,H,T,HD]
  _Float16* Kh  = (_Float16*)(ws + off); off += szX;              // [B,H,T,HD]
  _Float16* Vth = (_Float16*)(ws + off); off += szX;              // [B,H,HD,T]
  _Float16* Yh  = (_Float16*)(ws + off); off += szX;              // [BT,D]

  const int nX8 = (BT * D_EMB) / 8;
  const int nW8 = (D_EMB * D_EMB) / 8;
  cvt_f32_to_f16<<<(nX8 + 255) / 256, 256, 0, stream>>>(x,  xh,  nX8);
  cvt_f32_to_f16<<<(nW8 + 255) / 256, 256, 0, stream>>>(Wq, wqh, nW8);
  cvt_f32_to_f16<<<(nW8 + 255) / 256, 256, 0, stream>>>(Wk, wkh, nW8);
  cvt_f32_to_f16<<<(nW8 + 255) / 256, 256, 0, stream>>>(Wv, wvh, nW8);
  cvt_f32_to_f16<<<(nW8 + 255) / 256, 256, 0, stream>>>(Wp, wph, nW8);

  // 512 M-tiles * 64 N-tiles = 32768 waves -> 4096 blocks of 8 waves
  qkv_kernel<<<4096, 256, 0, stream>>>(xh, wqh, wkh, wvh, Qh, Kh, Vth);

  // 4*16*128 = 8192 waves -> 1024 blocks
  attn_kernel<<<1024, 256, 0, stream>>>(Qh, Kh, Vth, Yh);

  proj_kernel<<<4096, 256, 0, stream>>>(Yh, wph, bp, out);
}
